// RelativePositionalMultiHeadAttention_51238959841576
// MI455X (gfx1250) — matrix operand order repaired and checked by reading, not compiled, not run on hardware
//
#include <hip/hip_runtime.h>

// ---------------------------------------------------------------------------
// Transformer-XL style relative-position MHA for gfx1250 (MI455X).
// All matmuls via v_wmma_f32_16x16x32_bf16 (wave32, 16x16 tiles, K=32).
// ---------------------------------------------------------------------------

#define DEV __device__ __forceinline__

typedef __attribute__((ext_vector_type(16))) __bf16 v16bf;
typedef __attribute__((ext_vector_type(8)))  float  v8f;

static constexpr int BB = 8;
static constexpr int TT = 1024;
static constexpr int DD = 256;
static constexpr int HH = 8;
static constexpr int DH = 32;   // DD / HH

// 32-byte per-lane WMMA operand fragment (16 bf16 values).
union FragAB {
  v16bf v;
  unsigned short h[16];
  uint4 q[2];
};

DEV unsigned short f2bf(float x) {
  union { float f; unsigned u; } c; c.f = x;
  unsigned r = c.u + 0x7FFFu + ((c.u >> 16) & 1u);  // round-to-nearest-even
  return (unsigned short)(r >> 16);
}

DEV v8f wmma_bf16(const FragAB& a, const FragAB& b, v8f c) {
  return __builtin_amdgcn_wmma_f32_16x16x32_bf16(
      /*neg_a=*/false, a.v, /*neg_b=*/false, b.v,
      /*c_mod=*/(short)0, c, /*reuse_a=*/false, /*reuse_b=*/false);
}

// ---------------------------------------------------------------------------
// fp32 -> bf16 conversion
// ---------------------------------------------------------------------------
__global__ void cvt_bf16_kernel(const float* __restrict__ in,
                                unsigned short* __restrict__ out, int n) {
  int i = blockIdx.x * 256 + threadIdx.x;
  if (i < n) out[i] = f2bf(in[i]);
}

// ---------------------------------------------------------------------------
// Sinusoidal relative PE, bf16, rows r=0..2047 (r=2047 is pad, never stored).
// pe[r, 2m] = sin((r-1023)*w_m), pe[r, 2m+1] = cos(...), w_m = 10000^(-2m/D)
// ---------------------------------------------------------------------------
__global__ void pe_kernel(unsigned short* __restrict__ pe) {
  int i = blockIdx.x * 256 + threadIdx.x;   // 2048*256 total
  int r = i >> 8, c = i & 255;
  float pos  = (float)(r - (TT - 1));
  float freq = __expf(-(float)(c & ~1) * (9.210340371976184f / (float)DD));
  float ang  = pos * freq;
  float val  = (c & 1) ? __cosf(ang) : __sinf(ang);
  pe[i] = f2bf(val);
}

// ---------------------------------------------------------------------------
// Generic  C[M,256] = A[M,K](bf16) @ B[256,K](bf16)^T  (+bias), one wave per
// 16x16 tile.  Optional fp32 and/or bf16 outputs (bf16 used for P / AV).
// ---------------------------------------------------------------------------
__global__ __launch_bounds__(32) void gemm_bf16_kernel(
    const unsigned short* __restrict__ A, const unsigned short* __restrict__ Bw,
    const float* __restrict__ bias, float* __restrict__ outF,
    unsigned short* __restrict__ outB, int M, int K) {
  const int tm = blockIdx.x, tn = blockIdx.y;
  const int lane = threadIdx.x, half = lane >> 4, mi = lane & 15;
  const int row  = tm * 16 + mi;
  const int rc   = (row < M - 1) ? row : (M - 1);   // clamp ragged M (P: 2047)
  const int wcol = tn * 16 + mi;

  v8f acc = {};
  for (int k0 = 0; k0 < K; k0 += 32) {
    FragAB a, b;
    const unsigned short* ap = A + (size_t)rc * K + k0 + half * 16;
    a.q[0] = *(const uint4*)ap;                    // 16 contiguous K
    a.q[1] = *(const uint4*)(ap + 8);
    const unsigned short* bp = Bw + (size_t)wcol * K + k0 + half * 16;
    b.q[0] = *(const uint4*)bp;                    // 16 contiguous K
    b.q[1] = *(const uint4*)(bp + 8);
    __builtin_prefetch(ap + 32, 0, 1);
    __builtin_prefetch(bp + 32, 0, 1);
    acc = wmma_bf16(a, b, acc);
  }
#pragma unroll
  for (int r = 0; r < 8; ++r) {
    const int m = tm * 16 + r + 8 * half;          // C/D layout: lanes>=16 -> M+8
    if (m < M) {
      const int c = tn * 16 + mi;
      const float v = acc[r] + (bias ? bias[c] : 0.0f);
      const size_t o = (size_t)m * 256 + c;
      if (outF) outF[o] = v;
      if (outB) outB[o] = f2bf(v);
    }
  }
}

// ---------------------------------------------------------------------------
// QKV projection: q/k/v = X @ W^T + b, scattered head-major.
//   z=0: q -> qu=(q+u_bias), qv=(q+v_bias)  [B,H,T,DH] bf16
//   z=1: k -> k                              [B,H,T,DH] bf16
//   z=2: v -> vT (transposed)                [B,H,DH,T] bf16
// ---------------------------------------------------------------------------
__global__ __launch_bounds__(32) void qkv_kernel(
    const unsigned short* __restrict__ X,
    const unsigned short* __restrict__ Wq, const unsigned short* __restrict__ Wk,
    const unsigned short* __restrict__ Wv,
    const float* __restrict__ bq, const float* __restrict__ bk,
    const float* __restrict__ bv,
    const float* __restrict__ ub, const float* __restrict__ vb,
    unsigned short* __restrict__ quO, unsigned short* __restrict__ qvO,
    unsigned short* __restrict__ kO,  unsigned short* __restrict__ vTO) {
  const int tm = blockIdx.x, tn = blockIdx.y, which = blockIdx.z;
  const unsigned short* W = (which == 0) ? Wq : (which == 1) ? Wk : Wv;
  const float* bias       = (which == 0) ? bq : (which == 1) ? bk : bv;
  const int lane = threadIdx.x, half = lane >> 4, mi = lane & 15;
  const int row = tm * 16 + mi, wcol = tn * 16 + mi;

  v8f acc = {};
  for (int k0 = 0; k0 < DD; k0 += 32) {
    FragAB a, b;
    const unsigned short* ap = X + (size_t)row * DD + k0 + half * 16;
    a.q[0] = *(const uint4*)ap;
    a.q[1] = *(const uint4*)(ap + 8);
    const unsigned short* bp = W + (size_t)wcol * DD + k0 + half * 16;
    b.q[0] = *(const uint4*)bp;
    b.q[1] = *(const uint4*)(bp + 8);
    __builtin_prefetch(ap + 32, 0, 1);
    acc = wmma_bf16(a, b, acc);
  }
#pragma unroll
  for (int r = 0; r < 8; ++r) {
    const int t = tm * 16 + r + 8 * half;       // token index 0..8191
    const int c = tn * 16 + mi;                 // model dim 0..255
    const float val = acc[r] + bias[c];
    const int bb = t >> 10, ii = t & 1023, hh = c >> 5, dd = c & 31;
    const size_t hi = (((size_t)(bb * HH + hh)) * TT + ii) * DH + dd;
    if (which == 0) {
      quO[hi] = f2bf(val + ub[c]);
      qvO[hi] = f2bf(val + vb[c]);
    } else if (which == 1) {
      kO[hi] = f2bf(val);
    } else {
      vTO[(((size_t)(bb * HH + hh)) * DH + dd) * TT + ii] = f2bf(val);
    }
  }
}

// ---------------------------------------------------------------------------
// Fused attention per (b, h, 16-row block): scores (content + rel-pos, WMMA)
// -> LDS strip -> softmax -> attn store -> AV (WMMA) -> avB (bf16).
// 4 waves/block; wave w owns j-tiles w, w+4, w+8, ...
// ---------------------------------------------------------------------------
__global__ __launch_bounds__(128) void attn_kernel(
    const unsigned short* __restrict__ qu, const unsigned short* __restrict__ qv,
    const unsigned short* __restrict__ kk, const unsigned short* __restrict__ vT,
    const unsigned short* __restrict__ Pp,   // padded: row r stored at r+16
    float* __restrict__ attn_out, unsigned short* __restrict__ avB) {
  extern __shared__ char smem[];
  float* S    = (float*)smem;                               // [16][1024] 64KB
  float* Mav  = (float*)(smem + 16 * 1024 * 4);             // [4][16][32]  8KB
  float* red  = (float*)(smem + 16 * 1024 * 4 + 2048 * 4);  // [16][8]
  float* rmax = red + 128;                                  // [16]
  float* rsum = rmax + 16;                                  // [16]

  const int it = blockIdx.x, h = blockIdx.y, b = blockIdx.z;
  const int i0 = it * 16;
  const int tid = threadIdx.x, w = tid >> 5, lane = tid & 31;
  const int half = lane >> 4, mi = lane & 15;
  const size_t bh = (size_t)(b * HH + h);

  // Load per-row A fragments for (q+u) and (q+v): two b128 loads each.
  FragAB aU, aV;
  {
    const unsigned short* qb1 = qu + (bh * TT + i0 + mi) * DH;
    const unsigned short* qb2 = qv + (bh * TT + i0 + mi) * DH;
#pragma unroll
    for (int g = 0; g < 2; ++g) {
      aU.q[g] = *(const uint4*)(qb1 + g * 8 + half * 16);
      aV.q[g] = *(const uint4*)(qb2 + g * 8 + half * 16);
    }
  }

  float* msc = Mav + w * 512;   // this wave's 16x32 scratch

  // ---- score phase -------------------------------------------------------
  for (int t = 0; t < 16; ++t) {
    const int j0 = (w + 4 * t) * 16;

    // content = (q+u) . k  (one WMMA; K = DH = 32)
    FragAB bk;
    const unsigned short* kp = kk + (bh * TT + j0 + mi) * DH + half * 16;
    bk.q[0] = *(const uint4*)kp;
    bk.q[1] = *(const uint4*)(kp + 8);
    v8f c = {};
    c = wmma_bf16(aU, bk, c);

    // rel-pos: r(i,j) = i-j+T-1 spans 31 values -> 16x32 strip, 2 WMMAs
    const int rbase = i0 - j0 + (TT - 1) - 15;
    const unsigned short* pp = Pp + ((size_t)(rbase + 16 + mi)) * DD + h * DH + half * 16;
    FragAB b1, b2;
    b1.q[0] = *(const uint4*)pp;
    b1.q[1] = *(const uint4*)(pp + 8);
    const unsigned short* pp2 = pp + 16 * DD;
    b2.q[0] = *(const uint4*)pp2;
    b2.q[1] = *(const uint4*)(pp2 + 8);
    v8f m0 = {}, m1 = {};
    m0 = wmma_bf16(aV, b1, m0);
    m1 = wmma_bf16(aV, b2, m1);

#pragma unroll
    for (int r = 0; r < 8; ++r) {
      const int m = r + 8 * half;
      msc[m * 32 + mi]      = m0[r];
      msc[m * 32 + 16 + mi] = m1[r];
    }
    // gather diagonal 15+m-n and combine, scale by 1/sqrt(D) = 1/16
#pragma unroll
    for (int r = 0; r < 8; ++r) {
      const int m = r + 8 * half;
      const float pos = msc[m * 32 + (15 + m - mi)];
      S[m * 1024 + j0 + mi] = (c[r] + pos) * 0.0625f;
    }
  }
  __syncthreads();

  // ---- softmax over each of the 16 rows (8 threads / row) ---------------
  {
    const int row = tid >> 3, part = tid & 7;
    float* Sr = S + row * 1024 + part * 128;
    float mx = -3.4e38f;
    for (int cc = 0; cc < 128; ++cc) mx = fmaxf(mx, Sr[cc]);
    red[row * 8 + part] = mx;
    __syncthreads();
    if (part == 0) {
      float m2 = red[row * 8];
      for (int p = 1; p < 8; ++p) m2 = fmaxf(m2, red[row * 8 + p]);
      rmax[row] = m2;
    }
    __syncthreads();
    const float rm = rmax[row];
    float sum = 0.0f;
    for (int cc = 0; cc < 128; ++cc) {
      float e = __expf(Sr[cc] - rm);
      Sr[cc] = e;
      sum += e;
    }
    red[row * 8 + part] = sum;
    __syncthreads();
    if (part == 0) {
      float s2 = 0.0f;
      for (int p = 0; p < 8; ++p) s2 += red[row * 8 + p];
      rsum[row] = s2;
    }
    __syncthreads();
    const float inv = 1.0f / rsum[row];
    float* arow = attn_out + (bh * TT + i0 + row) * TT + part * 128;
    for (int cc = 0; cc < 128; ++cc) {
      const float a = Sr[cc] * inv;
      Sr[cc] = a;        // keep normalized probs for AV
      arow[cc] = a;      // mandatory attn output
    }
    __syncthreads();
  }

  // ---- AV: out16[16,32] = S(16x1024) @ V(1024x32) ------------------------
  v8f acc0 = {}, acc1 = {};
  const unsigned short* vbase = vT + bh * DH * TT;
  for (int t = 0; t < 16; ++t) {
    const int j0 = (w + 4 * t) * 16;
    FragAB aS;
#pragma unroll
    for (int e = 0; e < 16; ++e) {
      const int kidx = half * 16 + e;  // A-layout K index
      aS.h[e] = f2bf(S[mi * 1024 + j0 + kidx]);
    }
    const unsigned short* p1 = vbase + (size_t)mi * TT + j0 + half * 16;
    const unsigned short* p2 = vbase + (size_t)(16 + mi) * TT + j0 + half * 16;
    FragAB b1, b2;
    b1.q[0] = *(const uint4*)p1;  b1.q[1] = *(const uint4*)(p1 + 8);
    b2.q[0] = *(const uint4*)p2;  b2.q[1] = *(const uint4*)(p2 + 8);
    acc0 = wmma_bf16(aS, b1, acc0);   // d = 0..15
    acc1 = wmma_bf16(aS, b2, acc1);   // d = 16..31
  }
  {
    float* ap = Mav + w * 512;
#pragma unroll
    for (int r = 0; r < 8; ++r) {
      const int m = r + 8 * half;
      ap[m * 32 + mi]      = acc0[r];
      ap[m * 32 + 16 + mi] = acc1[r];
    }
  }
  __syncthreads();
#pragma unroll
  for (int e = 0; e < 4; ++e) {
    const int lin = tid + 128 * e;          // 0..511
    const int m = lin >> 5, d = lin & 31;
    const float s = Mav[m * 32 + d] + Mav[512 + m * 32 + d] +
                    Mav[1024 + m * 32 + d] + Mav[1536 + m * 32 + d];
    avB[((size_t)b * TT + i0 + m) * DD + h * DH + d] = f2bf(s);
  }
}

// ---------------------------------------------------------------------------
extern "C" void kernel_launch(void* const* d_in, const int* in_sizes, int n_in,
                              void* d_out, int out_size, void* d_ws, size_t ws_size,
                              hipStream_t stream) {
  (void)in_sizes; (void)n_in; (void)out_size; (void)ws_size;

  const float* query = (const float*)d_in[0];
  const float* Wq    = (const float*)d_in[1];
  const float* bq    = (const float*)d_in[2];
  const float* Wk    = (const float*)d_in[3];
  const float* bk    = (const float*)d_in[4];
  const float* Wv    = (const float*)d_in[5];
  const float* bv    = (const float*)d_in[6];
  const float* Wp    = (const float*)d_in[7];
  const float* Wo    = (const float*)d_in[8];
  const float* bo    = (const float*)d_in[9];
  const float* ub    = (const float*)d_in[10];
  const float* vb    = (const float*)d_in[11];

  float* out_main = (float*)d_out;                                 // [B,T,D]
  float* attn_out = out_main + (size_t)BB * TT * DD;               // [B,H,T,T]

  // ---- workspace carve (256B aligned) ----
  char* ws = (char*)d_ws;
  size_t off = 0;
  auto carve = [&](size_t bytes) -> char* {
    char* p = ws + off;
    off += (bytes + 255) & ~(size_t)255;
    return p;
  };
  unsigned short* Xbf   = (unsigned short*)carve((size_t)BB * TT * DD * 2);
  unsigned short* WqB   = (unsigned short*)carve((size_t)DD * DD * 2);
  unsigned short* WkB   = (unsigned short*)carve((size_t)DD * DD * 2);
  unsigned short* WvB   = (unsigned short*)carve((size_t)DD * DD * 2);
  unsigned short* WpB   = (unsigned short*)carve((size_t)DD * DD * 2);
  unsigned short* WoB   = (unsigned short*)carve((size_t)DD * DD * 2);
  unsigned short* quB   = (unsigned short*)carve((size_t)BB * TT * DD * 2);
  unsigned short* qvB   = (unsigned short*)carve((size_t)BB * TT * DD * 2);
  unsigned short* kB    = (unsigned short*)carve((size_t)BB * TT * DD * 2);
  unsigned short* vTB   = (unsigned short*)carve((size_t)BB * TT * DD * 2);
  unsigned short* peB   = (unsigned short*)carve((size_t)2048 * DD * 2);
  unsigned short* Ppad  = (unsigned short*)carve((size_t)2080 * DD * 2);  // 16-row pad each side
  unsigned short* avB   = (unsigned short*)carve((size_t)BB * TT * DD * 2);

  // 1) fp32 -> bf16 conversions
  {
    int nx = BB * TT * DD;
    cvt_bf16_kernel<<<(nx + 255) / 256, 256, 0, stream>>>(query, Xbf, nx);
    int nw = DD * DD;
    cvt_bf16_kernel<<<(nw + 255) / 256, 256, 0, stream>>>(Wq, WqB, nw);
    cvt_bf16_kernel<<<(nw + 255) / 256, 256, 0, stream>>>(Wk, WkB, nw);
    cvt_bf16_kernel<<<(nw + 255) / 256, 256, 0, stream>>>(Wv, WvB, nw);
    cvt_bf16_kernel<<<(nw + 255) / 256, 256, 0, stream>>>(Wp, WpB, nw);
    cvt_bf16_kernel<<<(nw + 255) / 256, 256, 0, stream>>>(Wo, WoB, nw);
  }

  // 2) sinusoidal PE (bf16)
  pe_kernel<<<2048, 256, 0, stream>>>(peB);

  // 3) P = pe @ Wp^T into zero-padded buffer (rows shifted by +16)
  hipMemsetAsync(Ppad, 0, (size_t)2080 * DD * 2, stream);
  {
    dim3 grid((2 * TT - 1 + 15) / 16, DD / 16);
    gemm_bf16_kernel<<<grid, 32, 0, stream>>>(peB, WpB, nullptr, nullptr,
                                              Ppad + 16 * DD, 2 * TT - 1, DD);
  }

  // 4) QKV projections (+u/v bias folding, head-major / transposed scatter)
  {
    dim3 grid((BB * TT) / 16, DD / 16, 3);
    qkv_kernel<<<grid, 32, 0, stream>>>(Xbf, WqB, WkB, WvB, bq, bk, bv, ub, vb,
                                        quB, qvB, kB, vTB);
  }

  // 5) fused attention: scores -> softmax -> attn store -> AV
  {
    dim3 grid(TT / 16, HH, BB);
    size_t shmem = 16 * 1024 * 4   // S
                 + 4 * 16 * 32 * 4 // Mav / msc / av-partials
                 + 16 * 8 * 4      // red
                 + 16 * 4 + 16 * 4; // rmax, rsum
    attn_kernel<<<grid, 128, shmem, stream>>>(quB, qvB, kB, vTB, Ppad,
                                              attn_out, avB);
  }

  // 6) out = AV @ Wo^T + bo  (fp32 output)
  {
    dim3 grid((BB * TT) / 16, DD / 16);
    gemm_bf16_kernel<<<grid, 32, 0, stream>>>(avB, WoB, bo, out_main, nullptr,
                                              BB * TT, DD);
  }
}
